// SwinTransformerBlock_18382460027449
// MI455X (gfx1250) — matrix operand order repaired and checked by reading, not compiled, not run on hardware
//
#include <hip/hip_runtime.h>
#include <hip/hip_bf16.h>
#include <stdint.h>

// ---------------------------------------------------------------------------
// Swin block on gfx1250: bf16 WMMA GEMMs (async-to-LDS double-buffered) +
// fused windowed attention, wave32.
// ---------------------------------------------------------------------------

typedef __bf16 bf16;
typedef __bf16 v16bf __attribute__((ext_vector_type(16)));
typedef float  v8f   __attribute__((ext_vector_type(8)));
typedef unsigned int u32x4 __attribute__((ext_vector_type(4)));

union Frag16 { v16bf v; u32x4 q[2]; };
union Acc8   { v8f v; float f[8]; };

#define C_      192
#define HID_    768
#define HEADS_  6
#define HD_     32
#define NW_     4096          // total windows: 4 * 32 * 32
#define NTOK_   262144        // 4 * 256 * 256  == NW_ * 64
#define SCALE_Q 0.17677669529663687f   // 32^-0.5

__device__ __forceinline__ int swin_region(int v) {
    // H=W=256, WIN=8, SHIFT=4 -> slice boundaries at 248 / 252
    return v < 248 ? 0 : (v < 252 ? 1 : 2);
}

// --- CDNA5 async copy: global -> LDS, 16B per lane, tracked on ASYNCcnt ----
__device__ __forceinline__ void async_ld_b128(unsigned lds_addr, unsigned goff,
                                              const void* base) {
    asm volatile("global_load_async_to_lds_b128 %0, %1, %2"
                 :
                 : "v"(lds_addr), "v"(goff),
                   "s"((unsigned long long)(uintptr_t)base)
                 : "memory");
}
__device__ __forceinline__ void wait_async0() {
    asm volatile("s_wait_asynccnt 0x0" ::: "memory");
}
// LDS byte offset = low 32 bits of the generic address (shared aperture)
__device__ __forceinline__ unsigned lds_off(const void* p) {
    return (unsigned)(uintptr_t)p;
}

// ---------------------------------------------------------------------------
// fp32 -> bf16 elementwise (weight conversion)
// ---------------------------------------------------------------------------
__global__ void f2bf_kernel(const float* __restrict__ in, bf16* __restrict__ out, int n) {
    int i = blockIdx.x * 256 + threadIdx.x;
    if (i < n) out[i] = (bf16)in[i];
}

// ---------------------------------------------------------------------------
// LN1 fused with roll(-4,-4) + 8x8 window partition.
// One wave per token (6 channels per lane), wave32 shuffle reduction.
// Output row order = win*64 + n  (GEMM A-matrix order for QKV).
// ---------------------------------------------------------------------------
__global__ __launch_bounds__(256)
void ln1_window_kernel(const float* __restrict__ x,
                       const float* __restrict__ g, const float* __restrict__ b,
                       bf16* __restrict__ out) {
    int warp = threadIdx.x >> 5, lane = threadIdx.x & 31;
    int token = blockIdx.x * 8 + warp;            // windowed token id
    int win = token >> 6, n = token & 63;
    int bb = win >> 10, wrc = win & 1023;
    int wr = wrc >> 5, wc = wrc & 31;
    int i = n >> 3, j = n & 7;
    int r = ((wr << 3) + i + 4) & 255;            // source row after roll(-4)
    int c = ((wc << 3) + j + 4) & 255;
    const float* p = x + (((size_t)bb << 16) + ((size_t)r << 8) + c) * C_;

    float v[6], s = 0.f, s2 = 0.f;
#pragma unroll
    for (int k = 0; k < 6; ++k) { v[k] = p[lane + 32 * k]; s += v[k]; s2 += v[k] * v[k]; }
#pragma unroll
    for (int m = 1; m < 32; m <<= 1) { s += __shfl_xor(s, m, 32); s2 += __shfl_xor(s2, m, 32); }
    float mu = s * (1.f / C_);
    float rstd = rsqrtf(s2 * (1.f / C_) - mu * mu + 1e-5f);
#pragma unroll
    for (int k = 0; k < 6; ++k) {
        int ch = lane + 32 * k;
        out[(size_t)token * C_ + ch] = (bf16)((v[k] - mu) * rstd * g[ch] + b[ch]);
    }
}

// ---------------------------------------------------------------------------
// LN2 over fp32 h (linear token order), bf16 out.
// ---------------------------------------------------------------------------
__global__ __launch_bounds__(256)
void ln2_kernel(const float* __restrict__ h,
                const float* __restrict__ g, const float* __restrict__ b,
                bf16* __restrict__ out) {
    int warp = threadIdx.x >> 5, lane = threadIdx.x & 31;
    int token = blockIdx.x * 8 + warp;
    const float* p = h + (size_t)token * C_;
    float v[6], s = 0.f, s2 = 0.f;
#pragma unroll
    for (int k = 0; k < 6; ++k) { v[k] = p[lane + 32 * k]; s += v[k]; s2 += v[k] * v[k]; }
#pragma unroll
    for (int m = 1; m < 32; m <<= 1) { s += __shfl_xor(s, m, 32); s2 += __shfl_xor(s2, m, 32); }
    float mu = s * (1.f / C_);
    float rstd = rsqrtf(s2 * (1.f / C_) - mu * mu + 1e-5f);
#pragma unroll
    for (int k = 0; k < 6; ++k) {
        int ch = lane + 32 * k;
        out[(size_t)token * C_ + ch] = (bf16)((v[k] - mu) * rstd * g[ch] + b[ch]);
    }
}

// ---------------------------------------------------------------------------
// Tiled bf16 WMMA GEMM:  D(M x Nout) = A(M x K) * W(Nout x K)^T + bias
// BM=128, BN=64, BK=32. 256 threads = 8 waves, wave grid 4x2, each wave a
// 32x32 output tile (2x2 WMMA 16x16 frags).
// Staging: double-buffered LDS tiles filled by GLOBAL_LOAD_ASYNC_TO_LDS_B128
// (3 instructions per wave per k-step), overlapped with WMMA on current tile.
// Epilogues by template arg:
//   0 = QKV scatter (+q scaling), 1 = proj + window-reverse + residual,
//   2 = FC1 + exact GELU, 3 = FC2 + residual -> fp32 out.
// ---------------------------------------------------------------------------
template <int EPI>
__global__ __launch_bounds__(256)
void gemm_wmma_kernel(const bf16* __restrict__ A, const bf16* __restrict__ W,
                      const float* __restrict__ bias, int K, int Nout,
                      bf16* __restrict__ outB, float* __restrict__ outF,
                      const float* __restrict__ addSrc) {
    __shared__ bf16 sA[2][128][40];   // +8 pad keeps 16B alignment, kills conflicts
    __shared__ bf16 sB[2][64][40];

    const int tid = threadIdx.x;
    const int m0 = blockIdx.x * 128;
    const int n0 = blockIdx.y * 64;
    const int w = tid >> 5, lane = tid & 31;
    const int wm = w >> 1, wn = w & 1;            // wave tile coords (4x2)
    const int nn = lane & 15, half = lane >> 4;

    Acc8 acc[2][2];
#pragma unroll
    for (int a = 0; a < 2; ++a)
#pragma unroll
        for (int bq = 0; bq < 2; ++bq)
#pragma unroll
            for (int r = 0; r < 8; ++r) acc[a][bq].f[r] = 0.f;

    // async staging: A tile = 512 16B-chunks (16 wave-instrs), B = 256 (8)
    auto issue_tile = [&](int kt, int bufsel) {
#pragma unroll
        for (int ii = 0; ii < 2; ++ii) {
            int chunk = (w + ii * 8) * 32 + lane;          // 0..511
            int row = chunk >> 2, col = (chunk & 3) * 8;   // 4 chunks per 64B row
            unsigned la = lds_off(&sA[bufsel][row][col]);
            unsigned go = (unsigned)(((size_t)(m0 + row) * K + kt * 32 + col) * 2);
            async_ld_b128(la, go, A);
        }
        {
            int chunk = w * 32 + lane;                     // 0..255
            int row = chunk >> 2, col = (chunk & 3) * 8;
            unsigned la = lds_off(&sB[bufsel][row][col]);
            unsigned go = (unsigned)(((size_t)(n0 + row) * K + kt * 32 + col) * 2);
            async_ld_b128(la, go, W);
        }
    };

    const int KT = K >> 5;
    issue_tile(0, 0);
    wait_async0();
    __syncthreads();

    for (int kt = 0; kt < KT; ++kt) {
        const int cur = kt & 1;
        if (kt + 1 < KT) issue_tile(kt + 1, cur ^ 1);   // overlap copy w/ compute

        Frag16 af[2], bf[2];
#pragma unroll
        for (int mt = 0; mt < 2; ++mt) {
            int row = wm * 32 + mt * 16 + nn;
            af[mt].q[0] = *(const u32x4*)&sA[cur][row][half * 8];        // K = kc*8..
            af[mt].q[1] = *(const u32x4*)&sA[cur][row][16 + half * 8];   // K = 16+kc*8..
        }
#pragma unroll
        for (int nt = 0; nt < 2; ++nt) {
            int col = wn * 32 + nt * 16 + nn;
            bf[nt].q[0] = *(const u32x4*)&sB[cur][col][half * 8];        // K = kc*8..
            bf[nt].q[1] = *(const u32x4*)&sB[cur][col][16 + half * 8];   // K = 16+kc*8..
        }
#pragma unroll
        for (int mt = 0; mt < 2; ++mt)
#pragma unroll
            for (int nt = 0; nt < 2; ++nt)
                acc[mt][nt].v = __builtin_amdgcn_wmma_f32_16x16x32_bf16(
                    false, af[mt].v, false, bf[nt].v, (short)0, acc[mt][nt].v, false, false);

        wait_async0();      // next tile landed (no-op on last iteration)
        __syncthreads();
    }

    // epilogue: lane holds D rows r + 8*half, col nn within each 16x16 tile
#pragma unroll
    for (int mt = 0; mt < 2; ++mt) {
#pragma unroll
        for (int nt = 0; nt < 2; ++nt) {
#pragma unroll
            for (int r = 0; r < 8; ++r) {
                int row = m0 + wm * 32 + mt * 16 + r + 8 * half;
                int col = n0 + wn * 32 + nt * 16 + nn;
                float v = acc[mt][nt].f[r] + bias[col];
                if (EPI == 0) {                      // QKV scatter
                    int t = col / C_;
                    int jj = col - t * C_;
                    int head = jj >> 5, d = jj & 31;
                    if (t == 0) v *= SCALE_Q;
                    size_t win = (size_t)(row >> 6);
                    int n = row & 63;
                    size_t dst = ((((size_t)t * NW_ + win) * HEADS_ + head) << 11)
                               + ((size_t)n << 5) + d;
                    outB[dst] = (bf16)v;
                } else if (EPI == 1) {               // proj: window-reverse + roll + residual
                    int win = row >> 6, n = row & 63;
                    int bb = win >> 10, wrc = win & 1023;
                    int wr = wrc >> 5, wc = wrc & 31;
                    int i = n >> 3, j = n & 7;
                    int rr = ((wr << 3) + i + 4) & 255;
                    int cc = ((wc << 3) + j + 4) & 255;
                    size_t tok = ((size_t)bb << 16) + ((size_t)rr << 8) + cc;
                    size_t o = tok * C_ + col;
                    outF[o] = addSrc[o] + v;
                } else if (EPI == 2) {               // FC1 + exact GELU
                    float gl = 0.5f * v * (1.f + erff(v * 0.70710678118654752f));
                    outB[(size_t)row * HID_ + col] = (bf16)gl;
                } else {                             // FC2 + residual -> final out
                    size_t o = (size_t)row * C_ + col;
                    outF[o] = addSrc[o] + v;
                }
            }
        }
    }
}

// ---------------------------------------------------------------------------
// Windowed attention: one block per (window, head), 128 threads = 4 waves.
// Each wave owns 16 query rows: S = q*k^T (4 WMMA), bias+mask, row softmax
// (shuffle reduce in 16-lane half-groups), P staged via LDS -> P*V (4 WMMA).
// ---------------------------------------------------------------------------
__global__ __launch_bounds__(128)
void attn_kernel(const bf16* __restrict__ qkv, const float* __restrict__ rpb,
                 bf16* __restrict__ outA) {
    __shared__ bf16 sQ[64][40];
    __shared__ bf16 sK[64][40];
    __shared__ bf16 sVT[32][72];     // V transposed: [d][key]
    __shared__ bf16 sP[4][16][72];   // per-wave softmax tile staging

    const int blk = blockIdx.x;
    const int head = blk % HEADS_;
    const int win = blk / HEADS_;
    const int tid = threadIdx.x;
    const size_t TSTR = (size_t)NW_ * HEADS_ * 2048;
    const size_t base = (((size_t)win * HEADS_) + head) << 11;

    {   // stage q, k, v (v transposed)
        int row = tid >> 1, seg = (tid & 1) * 16;
        *(u32x4*)&sQ[row][seg] = *(const u32x4*)(qkv + base + row * 32 + seg);
        *(u32x4*)&sK[row][seg] = *(const u32x4*)(qkv + TSTR + base + row * 32 + seg);
        const bf16* vs = qkv + 2 * TSTR + base + row * 32 + seg;
#pragma unroll
        for (int e = 0; e < 16; ++e) sVT[seg + e][row] = vs[e];
    }
    __syncthreads();

    const int w = tid >> 5, lane = tid & 31;
    const int nn = lane & 15, half = lane >> 4;

    // S = q * k^T, 16x64 per wave, K=32 in one WMMA step
    Frag16 qa;
    qa.q[0] = *(const u32x4*)&sQ[w * 16 + nn][half * 8];
    qa.q[1] = *(const u32x4*)&sQ[w * 16 + nn][16 + half * 8];
    Acc8 accS[4];
#pragma unroll
    for (int t = 0; t < 4; ++t) {
#pragma unroll
        for (int r = 0; r < 8; ++r) accS[t].f[r] = 0.f;
        Frag16 kb;
        kb.q[0] = *(const u32x4*)&sK[t * 16 + nn][half * 8];
        kb.q[1] = *(const u32x4*)&sK[t * 16 + nn][16 + half * 8];
        accS[t].v = __builtin_amdgcn_wmma_f32_16x16x32_bf16(
            false, qa.v, false, kb.v, (short)0, accS[t].v, false, false);
    }

    // relative-position bias + shifted-window mask (analytic)
    const int wrc = win & 1023;
    const int wr = wrc >> 5, wc = wrc & 31;
#pragma unroll
    for (int t = 0; t < 4; ++t) {
#pragma unroll
        for (int r = 0; r < 8; ++r) {
            int q = w * 16 + r + 8 * half;     // query index in window
            int kk = t * 16 + nn;              // key index in window
            int iq = q >> 3, jq = q & 7, ik = kk >> 3, jk = kk & 7;
            float bv = rpb[(((iq - ik + 7) * 15) + (jq - jk + 7)) * HEADS_ + head];
            int rq = (wr << 3) + iq, cq = (wc << 3) + jq;
            int rk = (wr << 3) + ik, ck = (wc << 3) + jk;
            int regq = swin_region(rq) * 3 + swin_region(cq);
            int regk = swin_region(rk) * 3 + swin_region(ck);
            accS[t].f[r] += bv + ((regq != regk) ? -100.f : 0.f);
        }
    }

    // row softmax: each row lives in one 16-lane half-group, 4 tiles x reg r
#pragma unroll
    for (int r = 0; r < 8; ++r) {
        float mx = fmaxf(fmaxf(accS[0].f[r], accS[1].f[r]),
                         fmaxf(accS[2].f[r], accS[3].f[r]));
#pragma unroll
        for (int m = 1; m < 16; m <<= 1) mx = fmaxf(mx, __shfl_xor(mx, m, 32));
        float sum = 0.f;
#pragma unroll
        for (int t = 0; t < 4; ++t) {
            float e = __expf(accS[t].f[r] - mx);
            accS[t].f[r] = e;
            sum += e;
        }
#pragma unroll
        for (int m = 1; m < 16; m <<= 1) sum += __shfl_xor(sum, m, 32);
        float inv = 1.f / sum;
#pragma unroll
        for (int t = 0; t < 4; ++t)
            sP[w][r + 8 * half][t * 16 + nn] = (bf16)(accS[t].f[r] * inv);
    }
    __syncthreads();

    // O = P * V : K=64 as two WMMA k-steps, two d-tiles of 16
    Acc8 accO[2];
#pragma unroll
    for (int dt = 0; dt < 2; ++dt)
#pragma unroll
        for (int r = 0; r < 8; ++r) accO[dt].f[r] = 0.f;
#pragma unroll
    for (int ks = 0; ks < 2; ++ks) {
        Frag16 pa;
        pa.q[0] = *(const u32x4*)&sP[w][nn][ks * 32 + half * 8];
        pa.q[1] = *(const u32x4*)&sP[w][nn][ks * 32 + 16 + half * 8];
#pragma unroll
        for (int dt = 0; dt < 2; ++dt) {
            Frag16 vb;
            vb.q[0] = *(const u32x4*)&sVT[dt * 16 + nn][ks * 32 + half * 8];
            vb.q[1] = *(const u32x4*)&sVT[dt * 16 + nn][ks * 32 + 16 + half * 8];
            accO[dt].v = __builtin_amdgcn_wmma_f32_16x16x32_bf16(
                false, pa.v, false, vb.v, (short)0, accO[dt].v, false, false);
        }
    }
#pragma unroll
    for (int dt = 0; dt < 2; ++dt)
#pragma unroll
        for (int r = 0; r < 8; ++r) {
            int q = w * 16 + r + 8 * half;
            int d = dt * 16 + nn;
            outA[((size_t)win * 64 + q) * C_ + head * HD_ + d] = (bf16)accO[dt].f[r];
        }
}

// ---------------------------------------------------------------------------
// Host launcher
// ---------------------------------------------------------------------------
extern "C" void kernel_launch(void* const* d_in, const int* in_sizes, int n_in,
                              void* d_out, int out_size, void* d_ws, size_t ws_size,
                              hipStream_t stream) {
    (void)in_sizes; (void)n_in; (void)out_size; (void)ws_size;
    const float* x      = (const float*)d_in[0];
    const float* qkv_w  = (const float*)d_in[1];
    const float* qkv_b  = (const float*)d_in[2];
    const float* proj_w = (const float*)d_in[3];
    const float* proj_b = (const float*)d_in[4];
    const float* rpb    = (const float*)d_in[5];
    const float* n1_w   = (const float*)d_in[6];
    const float* n1_b   = (const float*)d_in[7];
    const float* n2_w   = (const float*)d_in[8];
    const float* n2_b   = (const float*)d_in[9];
    const float* fc1_w  = (const float*)d_in[10];
    const float* fc1_b  = (const float*)d_in[11];
    const float* fc2_w  = (const float*)d_in[12];
    const float* fc2_b  = (const float*)d_in[13];
    float* out = (float*)d_out;

    // workspace layout (bytes)
    const size_t A_BYTES = (size_t)NTOK_ * C_ * 2;    // 100.7 MB: xw -> attn_out -> ln2m
    const size_t B_BYTES = (size_t)NTOK_ * HID_ * 2;  // 402.7 MB: qkv -> fc1out
    const size_t H_BYTES = (size_t)NTOK_ * C_ * 4;    // 201.3 MB: fp32 residual h
    char* ws = (char*)d_ws;
    bf16*  bufA = (bf16*)ws;
    bf16*  bufB = (bf16*)(ws + A_BYTES);
    float* hbuf = (float*)(ws + A_BYTES + B_BYTES);
    bf16*  wcv  = (bf16*)(ws + A_BYTES + B_BYTES + H_BYTES);
    bf16* qkv_wb  = wcv;                       // 576*192
    bf16* proj_wb = qkv_wb  + 576 * 192;       // 192*192
    bf16* fc1_wb  = proj_wb + 192 * 192;       // 768*192
    bf16* fc2_wb  = fc1_wb  + 768 * 192;       // 192*768

    // 1) weights fp32 -> bf16
    f2bf_kernel<<<(576 * 192 + 255) / 256, 256, 0, stream>>>(qkv_w, qkv_wb, 576 * 192);
    f2bf_kernel<<<(192 * 192 + 255) / 256, 256, 0, stream>>>(proj_w, proj_wb, 192 * 192);
    f2bf_kernel<<<(768 * 192 + 255) / 256, 256, 0, stream>>>(fc1_w, fc1_wb, 768 * 192);
    f2bf_kernel<<<(192 * 768 + 255) / 256, 256, 0, stream>>>(fc2_w, fc2_wb, 192 * 768);

    // 2) LN1 + roll + window partition -> bufA (windowed token order, bf16)
    ln1_window_kernel<<<NTOK_ / 8, 256, 0, stream>>>(x, n1_w, n1_b, bufA);

    // 3) QKV GEMM: (262144 x 192) * (576 x 192)^T -> scattered qkv in bufB
    gemm_wmma_kernel<0><<<dim3(NTOK_ / 128, 576 / 64), 256, 0, stream>>>(
        bufA, qkv_wb, qkv_b, C_, 576, bufB, nullptr, nullptr);

    // 4) attention per (window, head) -> attn_out in bufA
    attn_kernel<<<NW_ * HEADS_, 128, 0, stream>>>(bufB, rpb, bufA);

    // 5) proj GEMM + window-reverse + roll + residual -> fp32 h
    gemm_wmma_kernel<1><<<dim3(NTOK_ / 128, 192 / 64), 256, 0, stream>>>(
        bufA, proj_wb, proj_b, C_, C_, nullptr, hbuf, x);

    // 6) LN2 -> bufA (bf16)
    ln2_kernel<<<NTOK_ / 8, 256, 0, stream>>>(hbuf, n2_w, n2_b, bufA);

    // 7) FC1 + GELU -> bufB (bf16)
    gemm_wmma_kernel<2><<<dim3(NTOK_ / 128, HID_ / 64), 256, 0, stream>>>(
        bufA, fc1_wb, fc1_b, C_, HID_, bufB, nullptr, nullptr);

    // 8) FC2 + residual h -> final fp32 output
    gemm_wmma_kernel<3><<<dim3(NTOK_ / 128, 192 / 64), 256, 0, stream>>>(
        bufB, fc2_wb, fc2_b, HID_, C_, nullptr, out, hbuf);
}
